// BasicTransformerBlock_73108933312528
// MI455X (gfx1250) — matrix operand order repaired and checked by reading, not compiled, not run on hardware
//
#include <hip/hip_runtime.h>
#include <hip/hip_bf16.h>
#include <math.h>

// ---------------------------------------------------------------------------
// Shapes (compile-time constants from the reference)
//   B=8, L=1024, S=77, D=512, H=8, DH=64, CD=768, FF=2048, EPS=1e-5
//   Tokens TQ = 8192.  Cross keys padded per-batch to 80 -> TKc = 640.
// ---------------------------------------------------------------------------

typedef __bf16  bf16_t;
typedef bf16_t  bf16x16 __attribute__((ext_vector_type(16)));
typedef float   f32x8   __attribute__((ext_vector_type(8)));

union Frag { unsigned u[8]; bf16x16 b; };

__device__ __forceinline__ unsigned short f2bf(float f) {
  unsigned u = __float_as_uint(f);
  unsigned r = 0x7FFFu + ((u >> 16) & 1u);   // round-to-nearest-even
  return (unsigned short)((u + r) >> 16);
}

__device__ __forceinline__ f32x8 wmma_bf16(const Frag& a, const Frag& b, f32x8 c) {
  // 8 args: (neg_a, A, neg_b, B, c_mod, C, reuse_a, reuse_b)
  return __builtin_amdgcn_wmma_f32_16x16x32_bf16(false, a.b, false, b.b,
                                                 (short)0, c, false, false);
}

// ---------------------------------------------------------------------------
// Weight pack: f32 W[K][N] -> u32 P[K/2][N], each u32 = {bf16(W[2k+1][n]),bf16(W[2k][n])}
// Feeds this file's B-operand k order (contiguous, same as A): VGPR v holds K-pair (2v,2v+1)
// for lanes 0-15 and (16+2v,17+2v) for lanes 16-31 -> single b32 load per VGPR.
// ---------------------------------------------------------------------------
__global__ void pack_w_kernel(const float* __restrict__ W, unsigned* __restrict__ P,
                              int K, int N) {
  int idx = blockIdx.x * blockDim.x + threadIdx.x;
  int total = (K >> 1) * N;
  if (idx >= total) return;
  int n = idx % N, k2 = idx / N;
  unsigned lo = f2bf(W[(2 * k2) * N + n]);
  unsigned hi = f2bf(W[(2 * k2 + 1) * N + n]);
  P[idx] = lo | (hi << 16);
}

// context (8,77,768) f32 -> bf16 padded to (8,80,768), zero pad rows
__global__ void ctx_pad_kernel(const float* __restrict__ C, unsigned short* __restrict__ O) {
  int idx = blockIdx.x * blockDim.x + threadIdx.x;
  if (idx >= 640 * 768) return;
  int r = idx / 768, c = idx % 768;
  int b = r / 80, s = r % 80;
  float v = (s < 77) ? C[(b * 77 + s) * 768 + c] : 0.0f;
  O[idx] = f2bf(v);
}

// LayerNorm over D=512; one block (256 thr) per token; bf16 output
__global__ void __launch_bounds__(256, 1)
ln_kernel(const float* __restrict__ X, const float* __restrict__ g,
          const float* __restrict__ bt, unsigned short* __restrict__ H) {
  int row = blockIdx.x, tid = threadIdx.x;
  const float* x = X + (size_t)row * 512;
  float x0 = x[tid], x1 = x[tid + 256];
  float s = x0 + x1, q = x0 * x0 + x1 * x1;
  #pragma unroll
  for (int o = 16; o > 0; o >>= 1) { s += __shfl_xor(s, o); q += __shfl_xor(q, o); }
  __shared__ float ss[8], qq[8];
  int w = tid >> 5, ln = tid & 31;
  if (ln == 0) { ss[w] = s; qq[w] = q; }
  __syncthreads();
  float ts = 0.f, tq = 0.f;
  #pragma unroll
  for (int i = 0; i < 8; ++i) { ts += ss[i]; tq += qq[i]; }
  float mean = ts * (1.0f / 512.0f);
  float var  = tq * (1.0f / 512.0f) - mean * mean;
  float rs   = rsqrtf(var + 1e-5f);
  H[(size_t)row * 512 + tid]       = f2bf((x0 - mean) * rs * g[tid] + bt[tid]);
  H[(size_t)row * 512 + tid + 256] = f2bf((x1 - mean) * rs * g[tid + 256] + bt[tid + 256]);
}

// ---------------------------------------------------------------------------
// Generic WMMA GEMM: C[M,N] = A[M,K](bf16, pair-packed u32) * W(packed u32)
// One wave per 16x64 output tile. MODE 0: bf16 out. MODE 1: bf16 transposed
// out (for V^T). MODE 2: f32 out = resid + acc + bias (fused residual).
// ---------------------------------------------------------------------------
template <int MODE>
__global__ void __launch_bounds__(256, 1)
gemm_wmma(const unsigned* __restrict__ A2, int K,
          const unsigned* __restrict__ Bp, int NB,
          unsigned short* __restrict__ outb, float* __restrict__ outf,
          const float* __restrict__ resid, const float* __restrict__ bias,
          int M, int N, int outStride) {
  int lane = threadIdx.x & 31;
  int wid  = (blockIdx.x * blockDim.x + threadIdx.x) >> 5;
  int tilesN = N >> 6, tilesM = (M + 15) >> 4;
  if (wid >= tilesM * tilesN) return;
  int tm = wid / tilesN, tn = wid % tilesN;
  int half = lane >> 4, l15 = lane & 15;
  int mrow = tm * 16 + l15;
  bool mv = mrow < M;
  int K2 = K >> 1;
  f32x8 acc[4] = {};
  for (int k0 = 0; k0 < K; k0 += 32) {
    Frag fa;
    if (mv && (k0 + 32 < K))
      __builtin_prefetch(&A2[(size_t)mrow * K2 + ((k0 + 32) >> 1)], 0, 0);
    #pragma unroll
    for (int v = 0; v < 8; ++v) {
      // A k order used here (contiguous): VGPR v, lanes 0-15 K=2v, lanes 16-31 +16
      int kk = k0 + v * 2 + half * 16;
      fa.u[v] = mv ? A2[(size_t)mrow * K2 + (kk >> 1)] : 0u;
    }
    #pragma unroll
    for (int nt = 0; nt < 4; ++nt) {
      int col = tn * 64 + nt * 16 + l15;
      Frag fb;
      #pragma unroll
      for (int v = 0; v < 8; ++v)
        fb.u[v] = Bp[(size_t)((k0 >> 1) + v + half * 8) * NB + col];
      acc[nt] = wmma_bf16(fa, fb, acc[nt]);
    }
  }
  #pragma unroll
  for (int nt = 0; nt < 4; ++nt) {
    #pragma unroll
    for (int j = 0; j < 8; ++j) {
      int row = tm * 16 + j + half * 8;       // C layout: VGPR j -> M=j (+8 for hi half)
      if (row >= M) continue;
      int col = tn * 64 + nt * 16 + l15;
      float v = acc[nt][j];
      if (MODE == 0)      outb[(size_t)row * outStride + col] = f2bf(v);
      else if (MODE == 1) outb[(size_t)col * outStride + row] = f2bf(v);
      else                outf[(size_t)row * outStride + col] =
                              resid[(size_t)row * outStride + col] + v + bias[col];
    }
  }
}

// ---------------------------------------------------------------------------
// GeGLU GEMM: computes val = h@W1[:, :2048]+b1, gate = h@W1[:, 2048:]+b1',
// writes bf16 val*gelu(gate). 16x32 output tile per wave (2 val + 2 gate
// accumulators) to stay inside the VGPR budget -- no scratch spills.
// ---------------------------------------------------------------------------
__global__ void __launch_bounds__(256, 1)
gemm_geglu(const unsigned* __restrict__ A2, const unsigned* __restrict__ Bp,
           const float* __restrict__ b1, unsigned short* __restrict__ G,
           int M) {
  const int K = 512, NB = 4096;
  int lane = threadIdx.x & 31;
  int wid  = (blockIdx.x * blockDim.x + threadIdx.x) >> 5;
  int tilesN = 2048 >> 5;                // 64 tiles of 32 cols
  int tilesM = M >> 4;                   // 512
  if (wid >= tilesM * tilesN) return;
  int tm = wid / tilesN, tn = wid % tilesN;
  int half = lane >> 4, l15 = lane & 15;
  int mrow = tm * 16 + l15;
  f32x8 av[2] = {}, ag[2] = {};
  for (int k0 = 0; k0 < K; k0 += 32) {
    Frag fa;
    #pragma unroll
    for (int v = 0; v < 8; ++v) {
      int kk = k0 + v * 2 + half * 16;
      fa.u[v] = A2[(size_t)mrow * (K >> 1) + (kk >> 1)];
    }
    #pragma unroll
    for (int nt = 0; nt < 2; ++nt) {
      int col = tn * 32 + nt * 16 + l15;
      Frag fv;
      #pragma unroll
      for (int v = 0; v < 8; ++v)
        fv.u[v] = Bp[(size_t)((k0 >> 1) + v + half * 8) * NB + col];
      av[nt] = wmma_bf16(fa, fv, av[nt]);
      Frag fg;
      #pragma unroll
      for (int v = 0; v < 8; ++v)
        fg.u[v] = Bp[(size_t)((k0 >> 1) + v + half * 8) * NB + col + 2048];
      ag[nt] = wmma_bf16(fa, fg, ag[nt]);
    }
  }
  #pragma unroll
  for (int nt = 0; nt < 2; ++nt) {
    #pragma unroll
    for (int j = 0; j < 8; ++j) {
      int row = tm * 16 + j + half * 8;
      int col = tn * 32 + nt * 16 + l15;
      float vv = av[nt][j] + b1[col];
      float gg = ag[nt][j] + b1[col + 2048];
      float ge = 0.5f * gg * (1.0f + erff(gg * 0.70710678118654752f));  // exact GELU
      G[(size_t)row * 2048 + col] = f2bf(vv * ge);
    }
  }
}

// ---------------------------------------------------------------------------
// Flash attention: one wave per (batch, head, 16-query tile).
// QK^T via WMMA (2 per 32-key tile), online softmax (shfl_xor row reductions
// inside each 16-lane half of the wave32 C-layout), P restaged via LDS into
// the A-fragment layout, then P@V via WMMA against transposed V (contiguous
// token pairs -> single b32 loads).
// ---------------------------------------------------------------------------
__global__ void __launch_bounds__(256, 1)
attn_kernel(const unsigned* __restrict__ Q2, const unsigned* __restrict__ Kq,
            const unsigned* __restrict__ VT2, unsigned short* __restrict__ O,
            int kvLen, int kvStride, int TK) {
  __shared__ __align__(16) unsigned short pbuf[8][512];   // per-wave 16x32 P tile
  int lane = threadIdx.x & 31;
  int w    = threadIdx.x >> 5;
  int wid  = blockIdx.x * 8 + w;
  int qt   = wid & 63;
  int head = (wid >> 6) & 7;
  int b    = wid >> 9;
  int half = lane >> 4, l15 = lane & 15;
  int qrow = b * 1024 + qt * 16 + l15;

  Frag qf[2];                                     // Q tile 16x64 as two 16x32 A frags
  #pragma unroll
  for (int c = 0; c < 2; ++c)
    #pragma unroll
    for (int v = 0; v < 8; ++v) {
      int kk = c * 32 + v * 2 + half * 16;
      qf[c].u[v] = Q2[(size_t)qrow * 256 + ((head * 64 + kk) >> 1)];
    }

  float mrow[8], lrow[8];
  #pragma unroll
  for (int j = 0; j < 8; ++j) { mrow[j] = -1e30f; lrow[j] = 0.0f; }
  f32x8 acc[4] = {};
  int TK2 = TK >> 1;

  for (int key0 = 0; key0 < kvLen; key0 += 32) {
    f32x8 s0 = {}, s1 = {};
    #pragma unroll
    for (int c = 0; c < 2; ++c) {
      Frag fb0, fb1;
      #pragma unroll
      for (int v = 0; v < 8; ++v) {
        int du = head * 32 + c * 16 + v + half * 8;     // u32 index of dh-pair
        int k0i = key0 + l15, k1i = key0 + 16 + l15;
        fb0.u[v] = (k0i < kvLen) ? Kq[(size_t)(b * kvStride + k0i) * 256 + du] : 0u;
        fb1.u[v] = (k1i < kvLen) ? Kq[(size_t)(b * kvStride + k1i) * 256 + du] : 0u;
      }
      s0 = wmma_bf16(qf[c], fb0, s0);
      s1 = wmma_bf16(qf[c], fb1, s1);
    }
    bool v0 = (key0 + l15) < kvLen;
    bool v1 = (key0 + 16 + l15) < kvLen;
    #pragma unroll
    for (int j = 0; j < 8; ++j) {
      float a0 = v0 ? s0[j] * 0.125f : -1e30f;          // DH^-0.5 = 0.125
      float a1 = v1 ? s1[j] * 0.125f : -1e30f;
      float rmax = fmaxf(a0, a1);
      for (int o = 8; o > 0; o >>= 1) rmax = fmaxf(rmax, __shfl_xor(rmax, o));
      float mn = fmaxf(mrow[j], rmax);
      float sc = __expf(mrow[j] - mn);
      float p0 = __expf(a0 - mn);
      float p1 = __expf(a1 - mn);
      float rs = p0 + p1;
      for (int o = 8; o > 0; o >>= 1) rs += __shfl_xor(rs, o);
      lrow[j] = lrow[j] * sc + rs;
      mrow[j] = mn;
      #pragma unroll
      for (int nt = 0; nt < 4; ++nt) acc[nt][j] *= sc;
      int r = j + half * 8;
      pbuf[w][r * 32 + l15]      = f2bf(p0);
      pbuf[w][r * 32 + 16 + l15] = f2bf(p1);
    }
    __syncthreads();
    Frag pf;                                            // P as 16x32 A frag from LDS
    #pragma unroll
    for (int v = 0; v < 8; ++v) {
      int kk = v * 2 + half * 16;
      pf.u[v] = *(const unsigned*)&pbuf[w][l15 * 32 + kk];
    }
    #pragma unroll
    for (int nt = 0; nt < 4; ++nt) {
      Frag fv;
      #pragma unroll
      for (int v = 0; v < 8; ++v) {
        int t = key0 + 2 * v + 16 * half;               // token pair (t, t+1)
        fv.u[v] = (t < kvLen)
          ? VT2[(size_t)(head * 64 + nt * 16 + l15) * TK2 + ((b * kvStride + t) >> 1)]
          : 0u;
      }
      acc[nt] = wmma_bf16(pf, fv, acc[nt]);
    }
    __syncthreads();
  }

  #pragma unroll
  for (int nt = 0; nt < 4; ++nt)
    #pragma unroll
    for (int j = 0; j < 8; ++j) {
      int row = b * 1024 + qt * 16 + j + half * 8;
      int col = head * 64 + nt * 16 + l15;
      O[(size_t)row * 512 + col] = f2bf(acc[nt][j] / lrow[j]);
    }
}

// ---------------------------------------------------------------------------
extern "C" void kernel_launch(void* const* d_in, const int* in_sizes, int n_in,
                              void* d_out, int out_size, void* d_ws, size_t ws_size,
                              hipStream_t stream) {
  (void)in_sizes; (void)n_in; (void)out_size; (void)ws_size;
  const float* x     = (const float*)d_in[0];
  const float* ctx   = (const float*)d_in[1];
  const float* ln1_g = (const float*)d_in[2];
  const float* ln1_b = (const float*)d_in[3];
  const float* ln2_g = (const float*)d_in[4];
  const float* ln2_b = (const float*)d_in[5];
  const float* ln3_g = (const float*)d_in[6];
  const float* ln3_b = (const float*)d_in[7];
  const float* a1_wq = (const float*)d_in[8];
  const float* a1_wk = (const float*)d_in[9];
  const float* a1_wv = (const float*)d_in[10];
  const float* a1_wo = (const float*)d_in[11];
  const float* a1_bo = (const float*)d_in[12];
  const float* a2_wq = (const float*)d_in[13];
  const float* a2_wk = (const float*)d_in[14];
  const float* a2_wv = (const float*)d_in[15];
  const float* a2_wo = (const float*)d_in[16];
  const float* a2_bo = (const float*)d_in[17];
  const float* ff_w1 = (const float*)d_in[18];
  const float* ff_b1 = (const float*)d_in[19];
  const float* ff_w2 = (const float*)d_in[20];
  const float* ff_b2 = (const float*)d_in[21];

  char* ws = (char*)d_ws;
  size_t off = 0;
  auto alloc = [&](size_t bytes) -> void* {
    void* p = ws + off;
    off = (off + bytes + 255) & ~(size_t)255;
    return p;
  };

  // packed weights (u32 pairs): (K/2)*N u32 each
  unsigned* pWq1 = (unsigned*)alloc(256 * 512 * 4);
  unsigned* pWk1 = (unsigned*)alloc(256 * 512 * 4);
  unsigned* pWv1 = (unsigned*)alloc(256 * 512 * 4);
  unsigned* pWo1 = (unsigned*)alloc(256 * 512 * 4);
  unsigned* pWq2 = (unsigned*)alloc(256 * 512 * 4);
  unsigned* pWk2 = (unsigned*)alloc(384 * 512 * 4);
  unsigned* pWv2 = (unsigned*)alloc(384 * 512 * 4);
  unsigned* pWo2 = (unsigned*)alloc(256 * 512 * 4);
  unsigned* pW1  = (unsigned*)alloc(256 * 4096 * 4);
  unsigned* pW2  = (unsigned*)alloc(1024 * 512 * 4);
  unsigned short* ctxb = (unsigned short*)alloc(640 * 768 * 2);   // padded bf16 context
  float* xr            = (float*)alloc((size_t)8192 * 512 * 4);   // residual stream
  unsigned short* hb   = (unsigned short*)alloc((size_t)8192 * 512 * 2); // LN output
  // 32 MB block: Q/K/VT/O for attention, reused as FFN intermediate G
  unsigned short* Qb   = (unsigned short*)alloc((size_t)4 * 8192 * 512 * 2);
  unsigned short* Kb   = Qb + (size_t)8192 * 512;
  unsigned short* VTb  = Kb + (size_t)8192 * 512;
  unsigned short* Ob   = VTb + (size_t)8192 * 512;
  unsigned short* Gb   = Qb;   // 8192 x 2048 bf16 = 32 MB, reuses whole block

  auto packW = [&](const float* W, unsigned* P, int K, int N) {
    int total = (K / 2) * N;
    pack_w_kernel<<<(total + 255) / 256, 256, 0, stream>>>(W, P, K, N);
  };
  packW(a1_wq, pWq1, 512, 512);
  packW(a1_wk, pWk1, 512, 512);
  packW(a1_wv, pWv1, 512, 512);
  packW(a1_wo, pWo1, 512, 512);
  packW(a2_wq, pWq2, 512, 512);
  packW(a2_wk, pWk2, 768, 512);
  packW(a2_wv, pWv2, 768, 512);
  packW(a2_wo, pWo2, 512, 512);
  packW(ff_w1, pW1, 512, 4096);
  packW(ff_w2, pW2, 2048, 512);

  ctx_pad_kernel<<<(640 * 768 + 255) / 256, 256, 0, stream>>>(ctx, ctxb);
  hipMemcpyAsync(xr, x, (size_t)8192 * 512 * 4, hipMemcpyDeviceToDevice, stream);

  // ---- self-attention ----
  ln_kernel<<<8192, 256, 0, stream>>>(xr, ln1_g, ln1_b, hb);
  gemm_wmma<0><<<512, 256, 0, stream>>>((const unsigned*)hb, 512, pWq1, 512,
                                        Qb, nullptr, nullptr, nullptr, 8192, 512, 512);
  gemm_wmma<0><<<512, 256, 0, stream>>>((const unsigned*)hb, 512, pWk1, 512,
                                        Kb, nullptr, nullptr, nullptr, 8192, 512, 512);
  gemm_wmma<1><<<512, 256, 0, stream>>>((const unsigned*)hb, 512, pWv1, 512,
                                        VTb, nullptr, nullptr, nullptr, 8192, 512, 8192);
  attn_kernel<<<512, 256, 0, stream>>>((const unsigned*)Qb, (const unsigned*)Kb,
                                       (const unsigned*)VTb, Ob, 1024, 1024, 8192);
  gemm_wmma<2><<<512, 256, 0, stream>>>((const unsigned*)Ob, 512, pWo1, 512,
                                        nullptr, xr, xr, a1_bo, 8192, 512, 512);

  // ---- cross-attention ----
  ln_kernel<<<8192, 256, 0, stream>>>(xr, ln2_g, ln2_b, hb);
  gemm_wmma<0><<<512, 256, 0, stream>>>((const unsigned*)hb, 512, pWq2, 512,
                                        Qb, nullptr, nullptr, nullptr, 8192, 512, 512);
  gemm_wmma<0><<<40, 256, 0, stream>>>((const unsigned*)ctxb, 768, pWk2, 512,
                                       Kb, nullptr, nullptr, nullptr, 640, 512, 512);
  gemm_wmma<1><<<40, 256, 0, stream>>>((const unsigned*)ctxb, 768, pWv2, 512,
                                       VTb, nullptr, nullptr, nullptr, 640, 512, 640);
  attn_kernel<<<512, 256, 0, stream>>>((const unsigned*)Qb, (const unsigned*)Kb,
                                       (const unsigned*)VTb, Ob, 77, 80, 640);
  gemm_wmma<2><<<512, 256, 0, stream>>>((const unsigned*)Ob, 512, pWo2, 512,
                                        nullptr, xr, xr, a2_bo, 8192, 512, 512);

  // ---- GeGLU FFN ----
  ln_kernel<<<8192, 256, 0, stream>>>(xr, ln3_g, ln3_b, hb);
  gemm_geglu<<<4096, 256, 0, stream>>>((const unsigned*)hb, pW1, ff_b1, Gb, 8192);
  gemm_wmma<2><<<512, 256, 0, stream>>>((const unsigned*)Gb, 2048, pW2, 512,
                                        nullptr, (float*)d_out, xr, ff_b2, 8192, 512, 512);
}
